// MLAProjection_76836964926314
// MI455X (gfx1250) — matrix operand order repaired and checked by reading, not compiled, not run on hardware
//
#include <hip/hip_runtime.h>

#define E_N   65536
#define DIM_N 1024
#define LQ_N  512
#define ROWS  16      // E-rows per workgroup
#define CPAD  516     // f32 LDS row stride (512+4, bank-conflict pad)
#define BPAD  520     // bf16 LDS row stride (512+8)

typedef __attribute__((ext_vector_type(16))) __bf16 v16bf;
typedef __attribute__((ext_vector_type(8)))  __bf16 v8bf;
typedef __attribute__((ext_vector_type(8)))  float  v8f;

union FragAB { v16bf v; v8bf h[2]; };

static __device__ __forceinline__ v8f wmma_bf16(const FragAB& a, const FragAB& b, v8f c) {
    return __builtin_amdgcn_wmma_f32_16x16x32_bf16(false, a.v, false, b.v,
                                                   (short)0, c, false, false);
}

// pack 8 f32 -> 8 bf16 (for A fragments loaded from f32 activations)
static __device__ __forceinline__ v8bf cvt_v8(const float4 a, const float4 b) {
    v8bf r;
    r[0] = (__bf16)a.x; r[1] = (__bf16)a.y; r[2] = (__bf16)a.z; r[3] = (__bf16)a.w;
    r[4] = (__bf16)b.x; r[5] = (__bf16)b.y; r[6] = (__bf16)b.z; r[7] = (__bf16)b.w;
    return r;
}

static __device__ __forceinline__ void store_tile(float (*s)[CPAD], int n0, int l16, int half, v8f acc) {
    #pragma unroll
    for (int r = 0; r < 8; ++r) s[half * 8 + r][n0 + l16] = acc[r];   // D: row = r+8*half, col = n0+l16
}

// ---- phase-2 GEMM: k-outer, one LDS A-fragment feeds 4 WMMAs (4 n-tiles per wave) ----
// weight tile index = wtile0 + wave + 8j ; output staged at local tile wave + 8j
static __device__ __forceinline__ void gemm4_lds(const __bf16 (*s_cbf)[BPAD],
                                                 const __bf16* __restrict__ W, int Kdim,
                                                 int wave, int l16, int half,
                                                 float (*s_out)[CPAD], int wtile0) {
    v8f acc0 = {0.f,0.f,0.f,0.f,0.f,0.f,0.f,0.f};
    v8f acc1 = acc0, acc2 = acc0, acc3 = acc0;
    const __bf16* arow = s_cbf[l16];
    const __bf16* w0 = W + (size_t)((wtile0 + wave +  0) * 16 + l16) * Kdim;
    const __bf16* w1 = W + (size_t)((wtile0 + wave +  8) * 16 + l16) * Kdim;
    const __bf16* w2 = W + (size_t)((wtile0 + wave + 16) * 16 + l16) * Kdim;
    const __bf16* w3 = W + (size_t)((wtile0 + wave + 24) * 16 + l16) * Kdim;
    for (int k = 0; k < Kdim; k += 32) {
        FragAB fa;
        fa.h[0] = *(const v8bf*)(arow + k + half * 16);       // K = k + half*16 + 0..7
        fa.h[1] = *(const v8bf*)(arow + k + half * 16 + 8);   // K = k + half*16 + 8 + 0..7
        FragAB b0, b1, b2, b3;
        b0.h[0] = *(const v8bf*)(w0 + k + half * 16);  b0.h[1] = *(const v8bf*)(w0 + k + half * 16 + 8);
        b1.h[0] = *(const v8bf*)(w1 + k + half * 16);  b1.h[1] = *(const v8bf*)(w1 + k + half * 16 + 8);
        b2.h[0] = *(const v8bf*)(w2 + k + half * 16);  b2.h[1] = *(const v8bf*)(w2 + k + half * 16 + 8);
        b3.h[0] = *(const v8bf*)(w3 + k + half * 16);  b3.h[1] = *(const v8bf*)(w3 + k + half * 16 + 8);
        acc0 = wmma_bf16(fa, b0, acc0);
        acc1 = wmma_bf16(fa, b1, acc1);
        acc2 = wmma_bf16(fa, b2, acc2);
        acc3 = wmma_bf16(fa, b3, acc3);
    }
    store_tile(s_out, (wave +  0) * 16, l16, half, acc0);
    store_tile(s_out, (wave +  8) * 16, l16, half, acc1);
    store_tile(s_out, (wave + 16) * 16, l16, half, acc2);
    store_tile(s_out, (wave + 24) * 16, l16, half, acc3);
}

// ---- phase-1 GEMM: A from global f32 (cvt in regs), k-outer, 4 n-tiles per wave ----
template <int KDIM, bool SPLIT>
static __device__ __forceinline__ void gemm4_glob(const float* __restrict__ a_lo,
                                                  const float* __restrict__ a_hi,
                                                  const __bf16* __restrict__ W,
                                                  int wave, int l16, int half,
                                                  float (*s_out)[CPAD]) {
    v8f acc0 = {0.f,0.f,0.f,0.f,0.f,0.f,0.f,0.f};
    v8f acc1 = acc0, acc2 = acc0, acc3 = acc0;
    const __bf16* w0 = W + (size_t)((wave +  0) * 16 + l16) * KDIM;
    const __bf16* w1 = W + (size_t)((wave +  8) * 16 + l16) * KDIM;
    const __bf16* w2 = W + (size_t)((wave + 16) * 16 + l16) * KDIM;
    const __bf16* w3 = W + (size_t)((wave + 24) * 16 + l16) * KDIM;
    for (int k = 0; k < KDIM; k += 32) {
        const float* arow = (!SPLIT || k < DIM_N) ? a_lo : a_hi;
        const int kk = SPLIT ? (k & (DIM_N - 1)) : k;
        // prefetch the activation stream one 128B cacheline ahead (global_prefetch_b8)
        __builtin_prefetch(arow + kk + 32, 0, 3);
        FragAB fa;
        const float4 a0 = *(const float4*)(arow + kk + half * 16);
        const float4 a1 = *(const float4*)(arow + kk + half * 16 + 4);
        const float4 a2 = *(const float4*)(arow + kk + half * 16 + 8);
        const float4 a3 = *(const float4*)(arow + kk + half * 16 + 12);
        fa.h[0] = cvt_v8(a0, a1);
        fa.h[1] = cvt_v8(a2, a3);
        FragAB b0, b1, b2, b3;
        b0.h[0] = *(const v8bf*)(w0 + k + half * 16);  b0.h[1] = *(const v8bf*)(w0 + k + half * 16 + 8);
        b1.h[0] = *(const v8bf*)(w1 + k + half * 16);  b1.h[1] = *(const v8bf*)(w1 + k + half * 16 + 8);
        b2.h[0] = *(const v8bf*)(w2 + k + half * 16);  b2.h[1] = *(const v8bf*)(w2 + k + half * 16 + 8);
        b3.h[0] = *(const v8bf*)(w3 + k + half * 16);  b3.h[1] = *(const v8bf*)(w3 + k + half * 16 + 8);
        acc0 = wmma_bf16(fa, b0, acc0);
        acc1 = wmma_bf16(fa, b1, acc1);
        acc2 = wmma_bf16(fa, b2, acc2);
        acc3 = wmma_bf16(fa, b3, acc3);
    }
    store_tile(s_out, (wave +  0) * 16, l16, half, acc0);
    store_tile(s_out, (wave +  8) * 16, l16, half, acc1);
    store_tile(s_out, (wave + 16) * 16, l16, half, acc2);
    store_tile(s_out, (wave + 24) * 16, l16, half, acc3);
}

// ---------------- Q path: c_Q = LN(x_src @ Wqd^T); Q_nope; Q_rope(+rope) ----------------
__global__ __launch_bounds__(256) void qpath_kernel(
    const float* __restrict__ x_src, const float* __restrict__ t,
    const float* __restrict__ gamma, const float* __restrict__ beta,
    const float* __restrict__ lts,
    const __bf16* __restrict__ Wqd, const __bf16* __restrict__ Wqn, const __bf16* __restrict__ Wqr,
    float* __restrict__ outQn, float* __restrict__ outQr)
{
    __shared__ float  s_c[ROWS][CPAD];
    __shared__ __bf16 s_cbf[ROWS][BPAD];
    __shared__ float  s_psum[ROWS][16], s_psq[ROWS][16];
    __shared__ float  s_mean[ROWS], s_rstd[ROWS];
    __shared__ float  s_t[ROWS], s_invts[16];

    const int tid  = threadIdx.x;
    const int lane = tid & 31, wave = tid >> 5;
    const int l16  = lane & 15, half = lane >> 4;
    const int row0 = blockIdx.x * ROWS;

    if (tid < 16) { s_invts[tid] = __expf(-lts[tid]); s_t[tid] = t[row0 + tid]; }

    // Phase 1: c tile (16 x 512), K = 1024
    {
        const float* arow = x_src + (size_t)(row0 + l16) * DIM_N;
        gemm4_glob<DIM_N, false>(arow, arow, Wqd, wave, l16, half, s_c);
    }
    __syncthreads();

    // LayerNorm: 16 threads/row partial sums, finalize, normalize -> bf16 LDS
    {
        const int r = tid >> 4, part = tid & 15;
        float s = 0.f, sq = 0.f;
        const float* rowp = s_c[r] + part * 32;
        #pragma unroll
        for (int i = 0; i < 32; ++i) { const float v = rowp[i]; s += v; sq += v * v; }
        s_psum[r][part] = s; s_psq[r][part] = sq;
    }
    __syncthreads();
    if (tid < ROWS) {
        float s = 0.f, sq = 0.f;
        #pragma unroll
        for (int i = 0; i < 16; ++i) { s += s_psum[tid][i]; sq += s_psq[tid][i]; }
        const float mean = s * (1.0f / 512.0f);
        const float var  = sq * (1.0f / 512.0f) - mean * mean;
        s_mean[tid] = mean;
        s_rstd[tid] = rsqrtf(var + 1e-5f);
    }
    __syncthreads();
    for (int i = 0; i < 32; ++i) {
        const int idx = tid + i * 256, r = idx >> 9, c = idx & 511;
        const float v = (s_c[r][c] - s_mean[r]) * s_rstd[r] * gamma[c] + beta[c];
        s_cbf[r][c] = (__bf16)v;
    }
    __syncthreads();

    // Phase 2a: Q_nope
    gemm4_lds(s_cbf, Wqn, LQ_N, wave, l16, half, s_c, 0);
    __syncthreads();
    for (int i = 0; i < 32; ++i) {
        const int idx = tid + i * 256, r = idx >> 9, c = idx & 511;
        outQn[(size_t)(row0 + r) * 512 + c] = s_c[r][c];
    }
    __syncthreads();

    // Phase 2b: Q_rope + RoPE epilogue (even/odd pairs via LDS; cross-lane in D layout)
    gemm4_lds(s_cbf, Wqr, LQ_N, wave, l16, half, s_c, 0);
    __syncthreads();
    for (int i = 0; i < 32; ++i) {
        const int idx = tid + i * 256, r = idx >> 9, c = idx & 511;
        const int d = c & 31, j = d >> 1;
        const float ang = s_t[r] * s_invts[j];
        const float ca = __cosf(ang), sa = __sinf(ang);
        const float e = s_c[r][c & ~1], o = s_c[r][c | 1];
        outQr[(size_t)(row0 + r) * 512 + c] = (d & 1) ? (e * sa + o * ca) : (e * ca - o * sa);
    }
}

// -------- KV path: c_KV = LN([x_dst|edge] @ Wkvd^T); K_nope; K_rope(+rope); V --------
__global__ __launch_bounds__(256) void kvpath_kernel(
    const float* __restrict__ x_dst, const float* __restrict__ edge, const float* __restrict__ t,
    const float* __restrict__ gamma, const float* __restrict__ beta,
    const float* __restrict__ lts,
    const __bf16* __restrict__ Wkvd, const __bf16* __restrict__ Wkn,
    const __bf16* __restrict__ Wkr,  const __bf16* __restrict__ Wv,
    float* __restrict__ outKn, float* __restrict__ outKr, float* __restrict__ outV)
{
    __shared__ float  s_c[ROWS][CPAD];
    __shared__ __bf16 s_cbf[ROWS][BPAD];
    __shared__ float  s_psum[ROWS][16], s_psq[ROWS][16];
    __shared__ float  s_mean[ROWS], s_rstd[ROWS];
    __shared__ float  s_t[ROWS], s_invts[16];

    const int tid  = threadIdx.x;
    const int lane = tid & 31, wave = tid >> 5;
    const int l16  = lane & 15, half = lane >> 4;
    const int row0 = blockIdx.x * ROWS;

    if (tid < 16) { s_invts[tid] = __expf(-lts[tid]); s_t[tid] = t[row0 + tid]; }

    // Phase 1: K = 2048, A = concat(x_dst, edge) (k-steps never straddle the 1024 seam)
    {
        const float* a_lo = x_dst + (size_t)(row0 + l16) * DIM_N;
        const float* a_hi = edge  + (size_t)(row0 + l16) * DIM_N;
        gemm4_glob<2 * DIM_N, true>(a_lo, a_hi, Wkvd, wave, l16, half, s_c);
    }
    __syncthreads();

    {
        const int r = tid >> 4, part = tid & 15;
        float s = 0.f, sq = 0.f;
        const float* rowp = s_c[r] + part * 32;
        #pragma unroll
        for (int i = 0; i < 32; ++i) { const float v = rowp[i]; s += v; sq += v * v; }
        s_psum[r][part] = s; s_psq[r][part] = sq;
    }
    __syncthreads();
    if (tid < ROWS) {
        float s = 0.f, sq = 0.f;
        #pragma unroll
        for (int i = 0; i < 16; ++i) { s += s_psum[tid][i]; sq += s_psq[tid][i]; }
        const float mean = s * (1.0f / 512.0f);
        const float var  = sq * (1.0f / 512.0f) - mean * mean;
        s_mean[tid] = mean;
        s_rstd[tid] = rsqrtf(var + 1e-5f);
    }
    __syncthreads();
    for (int i = 0; i < 32; ++i) {
        const int idx = tid + i * 256, r = idx >> 9, c = idx & 511;
        const float v = (s_c[r][c] - s_mean[r]) * s_rstd[r] * gamma[c] + beta[c];
        s_cbf[r][c] = (__bf16)v;
    }
    __syncthreads();

    // K_nope
    gemm4_lds(s_cbf, Wkn, LQ_N, wave, l16, half, s_c, 0);
    __syncthreads();
    for (int i = 0; i < 32; ++i) {
        const int idx = tid + i * 256, r = idx >> 9, c = idx & 511;
        outKn[(size_t)(row0 + r) * 512 + c] = s_c[r][c];
    }
    __syncthreads();

    // K_rope + RoPE
    gemm4_lds(s_cbf, Wkr, LQ_N, wave, l16, half, s_c, 0);
    __syncthreads();
    for (int i = 0; i < 32; ++i) {
        const int idx = tid + i * 256, r = idx >> 9, c = idx & 511;
        const int d = c & 31, j = d >> 1;
        const float ang = s_t[r] * s_invts[j];
        const float ca = __cosf(ang), sa = __sinf(ang);
        const float e = s_c[r][c & ~1], o = s_c[r][c | 1];
        outKr[(size_t)(row0 + r) * 512 + c] = (d & 1) ? (e * sa + o * ca) : (e * ca - o * sa);
    }
    __syncthreads();

    // V: N = 1024, done as two 512-col halves through the LDS staging buffer
    for (int hh = 0; hh < 2; ++hh) {
        gemm4_lds(s_cbf, Wv, LQ_N, wave, l16, half, s_c, hh * 32);
        __syncthreads();
        for (int i = 0; i < 32; ++i) {
            const int idx = tid + i * 256, r = idx >> 9, c = idx & 511;
            outV[(size_t)(row0 + r) * 1024 + hh * 512 + c] = s_c[r][c];
        }
        __syncthreads();
    }
}

// ---------------- weight f32 -> bf16 (weights are tiny: 6 MB total, L2-resident) ----------------
__global__ void cvt_bf16_kernel(const float* __restrict__ src, __bf16* __restrict__ dst, int n) {
    const int i = blockIdx.x * 256 + threadIdx.x;
    if (i < n) dst[i] = (__bf16)src[i];
}

extern "C" void kernel_launch(void* const* d_in, const int* in_sizes, int n_in,
                              void* d_out, int out_size, void* d_ws, size_t ws_size,
                              hipStream_t stream) {
    const float* x_src   = (const float*)d_in[0];
    const float* x_dst   = (const float*)d_in[1];
    const float* edge    = (const float*)d_in[2];
    const float* t       = (const float*)d_in[3];
    const float* W_qd    = (const float*)d_in[4];
    const float* q_gamma = (const float*)d_in[5];
    const float* q_beta  = (const float*)d_in[6];
    const float* W_qn    = (const float*)d_in[7];
    const float* W_qr    = (const float*)d_in[8];
    const float* W_kvd   = (const float*)d_in[9];
    const float* kv_gamma= (const float*)d_in[10];
    const float* kv_beta = (const float*)d_in[11];
    const float* W_kn    = (const float*)d_in[12];
    const float* W_kr    = (const float*)d_in[13];
    const float* W_v     = (const float*)d_in[14];
    const float* q_lts   = (const float*)d_in[15];
    const float* k_lts   = (const float*)d_in[16];

    __bf16* ws = (__bf16*)d_ws;
    const int nQD  = 512 * 1024;
    const int nSQ  = 512 * 512;
    const int nKVD = 512 * 2048;
    const int nV   = 1024 * 512;
    __bf16* bWqd  = ws;                 ws += nQD;
    __bf16* bWqn  = ws;                 ws += nSQ;
    __bf16* bWqr  = ws;                 ws += nSQ;
    __bf16* bWkvd = ws;                 ws += nKVD;
    __bf16* bWkn  = ws;                 ws += nSQ;
    __bf16* bWkr  = ws;                 ws += nSQ;
    __bf16* bWv   = ws;                 ws += nV;

    auto cvt = [&](const float* s, __bf16* d, int n) {
        cvt_bf16_kernel<<<(n + 255) / 256, 256, 0, stream>>>(s, d, n);
    };
    cvt(W_qd,  bWqd,  nQD);
    cvt(W_qn,  bWqn,  nSQ);
    cvt(W_qr,  bWqr,  nSQ);
    cvt(W_kvd, bWkvd, nKVD);
    cvt(W_kn,  bWkn,  nSQ);
    cvt(W_kr,  bWkr,  nSQ);
    cvt(W_v,   bWv,   nV);

    float* out   = (float*)d_out;
    float* outQn = out;
    float* outQr = out + (size_t)E_N * 512;
    float* outKn = out + (size_t)E_N * 512 * 2;
    float* outKr = out + (size_t)E_N * 512 * 3;
    float* outV  = out + (size_t)E_N * 512 * 4;

    qpath_kernel<<<E_N / ROWS, 256, 0, stream>>>(
        x_src, t, q_gamma, q_beta, q_lts, bWqd, bWqn, bWqr, outQn, outQr);
    kvpath_kernel<<<E_N / ROWS, 256, 0, stream>>>(
        x_dst, edge, t, kv_gamma, kv_beta, k_lts, bWkvd, bWkn, bWkr, bWv, outKn, outKr, outV);
}
